// EncoderBlock_25640954757295
// MI455X (gfx1250) — matrix operand order repaired and checked by reading, not compiled, not run on hardware
//
#include <hip/hip_runtime.h>
#include <hip/hip_bf16.h>

// ---- problem constants (reference is fixed-shape) ----
#define DM 1024   // d_model
#define SQ 2048   // sequence length
#define BB 2      // batch
#define NH 16     // heads
#define DK 64     // head dim
#define FF 4096   // d_ff
#define MT 4096   // BB*SQ tokens

typedef __attribute__((ext_vector_type(16))) __bf16 v16bf;
typedef __attribute__((ext_vector_type(8)))  float  v8f;

union Frag {
    uint4 u4[2];
    v16bf v;
};

__device__ __forceinline__ unsigned short f32_to_bf16u(float f) {
    unsigned int u = __float_as_uint(f);
    u += 0x7FFFu + ((u >> 16) & 1u);
    return (unsigned short)(u >> 16);
}

// ---------------------------------------------------------------------------
// LayerNorm (scalar alpha/bias, unbiased std, eps on std) -> bf16 activations
// one 256-thread block per token row
// ---------------------------------------------------------------------------
__global__ __launch_bounds__(256) void ln_kernel(const float* __restrict__ x,
                                                 const float* __restrict__ alpha,
                                                 const float* __restrict__ bias,
                                                 unsigned short* __restrict__ out) {
    __shared__ float red[256];
    const int row = blockIdx.x;
    const int tid = threadIdx.x;
    const float* xr = x + (size_t)row * DM;
    float v0 = xr[tid], v1 = xr[tid + 256], v2 = xr[tid + 512], v3 = xr[tid + 768];
    red[tid] = v0 + v1 + v2 + v3;
    __syncthreads();
    for (int off = 128; off > 0; off >>= 1) {
        if (tid < off) red[tid] += red[tid + off];
        __syncthreads();
    }
    const float mean = red[0] * (1.0f / DM);
    __syncthreads();
    const float d0 = v0 - mean, d1 = v1 - mean, d2 = v2 - mean, d3 = v3 - mean;
    red[tid] = d0 * d0 + d1 * d1 + d2 * d2 + d3 * d3;
    __syncthreads();
    for (int off = 128; off > 0; off >>= 1) {
        if (tid < off) red[tid] += red[tid + off];
        __syncthreads();
    }
    const float var = red[0] * (1.0f / (DM - 1));      // ddof=1
    const float inv = alpha[0] / (sqrtf(var) + 1e-4f); // eps on std
    const float bb  = bias[0];
    unsigned short* o = out + (size_t)row * DM;
    o[tid]       = f32_to_bf16u(d0 * inv + bb);
    o[tid + 256] = f32_to_bf16u(d1 * inv + bb);
    o[tid + 512] = f32_to_bf16u(d2 * inv + bb);
    o[tid + 768] = f32_to_bf16u(d3 * inv + bb);
}

// ---------------------------------------------------------------------------
// Weight convert fp32 [K,N] -> bf16 transposed [N,K] so GEMM B-fragments are
// contiguous along K (matches WMMA B-matrix VGPR layout).
// ---------------------------------------------------------------------------
__global__ __launch_bounds__(256) void wtrans_kernel(const float* __restrict__ W,
                                                     unsigned short* __restrict__ WT,
                                                     int K, int N, int kshift) {
    const int idx = blockIdx.x * 256 + threadIdx.x;   // over N*K
    const int n   = idx >> kshift;
    const int kk  = idx & (K - 1);
    WT[idx] = f32_to_bf16u(W[(size_t)kk * N + n]);
}

// ---------------------------------------------------------------------------
// mask (int, [B,S]) -> additive fp32 bias (0 or -1e9)
// ---------------------------------------------------------------------------
__global__ __launch_bounds__(256) void maskbias_kernel(const int* __restrict__ mask,
                                                       float* __restrict__ mb) {
    const int idx = blockIdx.x * 256 + threadIdx.x;
    mb[idx] = (mask[idx] == 0) ? -1e9f : 0.0f;
}

// ---------------------------------------------------------------------------
// GEMM: C[M,N] = A[M,K](bf16,row) * BT[N,K](bf16,row, i.e. B transposed) + bias
// 256 threads = 8 waves; block tile 128x128; wave tile 32x64 (2x4 WMMA frags).
// 8 WMMAs per 12 b128 loads per k-step. Epilogue variants select output.
// ---------------------------------------------------------------------------
enum { EPI_BF16 = 0, EPI_BF16_RELU = 1, EPI_F32_RES = 2, EPI_VT = 3 };

template <int EPI>
__global__ __launch_bounds__(256) void gemm_kernel(const unsigned short* __restrict__ A,
                                                   const unsigned short* __restrict__ BT,
                                                   const float* __restrict__ bias,
                                                   const float* __restrict__ resid,
                                                   void* __restrict__ out,
                                                   int N, int K) {
    const int tid   = threadIdx.x;
    const int lane  = tid & 31;
    const int wave  = tid >> 5;
    const int wr    = wave >> 1, wc = wave & 1;
    const int mbase = blockIdx.y * 128 + wr * 32;
    const int nbase = blockIdx.x * 128 + wc * 64;
    const int l16   = lane & 15;
    const int hi    = lane >> 4;

    const v8f zero = {0.f, 0.f, 0.f, 0.f, 0.f, 0.f, 0.f, 0.f};
    v8f acc[2][4];
#pragma unroll
    for (int i = 0; i < 2; ++i)
#pragma unroll
        for (int j = 0; j < 4; ++j) acc[i][j] = zero;

    const int koffA = hi * 8;   // A lane K offset (ISA 16-bit A layout)
    const int koffB = hi * 8;   // B lane K offset (same K order as A)

    for (int k0 = 0; k0 < K; k0 += 32) {
        Frag af[2], bf[4];
#pragma unroll
        for (int mt = 0; mt < 2; ++mt) {
            const unsigned short* pa = A + (size_t)(mbase + mt * 16 + l16) * K + k0 + koffA;
            af[mt].u4[0] = *(const uint4*)(pa);       // K = koffA .. +7
            af[mt].u4[1] = *(const uint4*)(pa + 16);  // K = koffA+16 .. +23
        }
#pragma unroll
        for (int nt = 0; nt < 4; ++nt) {
            const unsigned short* pb = BT + (size_t)(nbase + nt * 16 + l16) * K + k0 + koffB;
            bf[nt].u4[0] = *(const uint4*)(pb);      // K = koffB .. +7
            bf[nt].u4[1] = *(const uint4*)(pb + 16); // K = koffB+16 .. +23
        }
#pragma unroll
        for (int mt = 0; mt < 2; ++mt)
#pragma unroll
            for (int nt = 0; nt < 4; ++nt)
                acc[mt][nt] = __builtin_amdgcn_wmma_f32_16x16x32_bf16(
                    false, af[mt].v, false, bf[nt].v, (short)0, acc[mt][nt], false, false);
    }

#pragma unroll
    for (int mt = 0; mt < 2; ++mt) {
#pragma unroll
        for (int nt = 0; nt < 4; ++nt) {
            const int n  = nbase + nt * 16 + l16;
            const float bv = bias[n];
#pragma unroll
            for (int r = 0; r < 8; ++r) {
                const int m  = mbase + mt * 16 + hi * 8 + r;
                float val    = acc[mt][nt][r] + bv;
                if (EPI == EPI_BF16) {
                    ((unsigned short*)out)[(size_t)m * N + n] = f32_to_bf16u(val);
                } else if (EPI == EPI_BF16_RELU) {
                    ((unsigned short*)out)[(size_t)m * N + n] = f32_to_bf16u(fmaxf(val, 0.f));
                } else if (EPI == EPI_F32_RES) {
                    ((float*)out)[(size_t)m * N + n] = val + resid[(size_t)m * N + n];
                } else {  // EPI_VT: write V as V^T [B,H,DK,SQ] for attention B-fragments
                    const int hh = n >> 6, d = n & 63;
                    const int b = m >> 11, s = m & (SQ - 1);
                    ((unsigned short*)out)[(size_t)((b * NH + hh) * DK + d) * SQ + s] =
                        f32_to_bf16u(val);
                }
            }
        }
    }
}

// ---------------------------------------------------------------------------
// Flash-style attention. One wave per (b, h, 32-query block): two 16-row query
// tiles share each K/V fragment load (16 WMMAs per 16 b128 loads per 32-key
// chunk). Scores computed transposed (keys x queries) so softmax rows stay
// lane-local; P re-shaped via 2KB LDS/wave into the WMMA A-fragment layout.
// ---------------------------------------------------------------------------
__global__ __launch_bounds__(128) void attn_kernel(const unsigned short* __restrict__ Q,
                                                   const unsigned short* __restrict__ Kb,
                                                   const unsigned short* __restrict__ VT,
                                                   const float* __restrict__ maskb,
                                                   unsigned short* __restrict__ ctx) {
    __shared__ unsigned int plds[4 * 512];  // 4 waves * 2 q-tiles * (16 rows * 32 bf16)
    const int tid  = threadIdx.x;
    const int lane = tid & 31;
    const int wave = tid >> 5;
    const int wid  = blockIdx.x * 4 + wave;
    const int qt   = wid & (SQ / 32 - 1);       // 32-query block
    const int h    = (wid >> 6) & (NH - 1);
    const int b    = wid >> 10;
    const int l16  = lane & 15;
    const int hi   = lane >> 4;

    unsigned int* pw = plds + wave * 512;

    // Q^T fragments (B operand, d x q): [q-tile][dk-half]
    Frag qf[2][2];
#pragma unroll
    for (int qi = 0; qi < 2; ++qi) {
        const int qrow = qt * 32 + qi * 16 + l16;
        const unsigned short* qp = Q + (size_t)(b * SQ + qrow) * DM + h * DK;
#pragma unroll
        for (int hd = 0; hd < 2; ++hd) {
            const unsigned short* p = qp + hd * 32 + hi * 8;
            qf[qi][hd].u4[0] = *(const uint4*)(p);
            qf[qi][hd].u4[1] = *(const uint4*)(p + 16);
        }
    }

    const v8f zero = {0.f, 0.f, 0.f, 0.f, 0.f, 0.f, 0.f, 0.f};
    v8f o[2][4];
#pragma unroll
    for (int qi = 0; qi < 2; ++qi)
#pragma unroll
        for (int nt = 0; nt < 4; ++nt) o[qi][nt] = zero;
    float mrun[2] = {-1e30f, -1e30f};
    float lsum[2] = {0.f, 0.f};

    const unsigned short* kptr = Kb + (size_t)b * SQ * DM + h * DK;
    const unsigned short* vptr = VT + (size_t)(b * NH + h) * DK * SQ;
    const float* mrow = maskb + b * SQ;

    for (int key0 = 0; key0 < SQ; key0 += 32) {
        // K fragments, shared by both q-tiles: [key-tile][dk-half]
        Frag kf[2][2];
#pragma unroll
        for (int kt = 0; kt < 2; ++kt) {
            const unsigned short* kp = kptr + (size_t)(key0 + kt * 16 + l16) * DM;
#pragma unroll
            for (int hd = 0; hd < 2; ++hd) {
                const unsigned short* p = kp + hd * 32 + hi * 8;
                kf[kt][hd].u4[0] = *(const uint4*)(p);
                kf[kt][hd].u4[1] = *(const uint4*)(p + 16);
            }
        }
        // scores^T = K_tile (16xdk) x Q^T (dk x 16): 8 WMMAs
        v8f st[2][2];
#pragma unroll
        for (int qi = 0; qi < 2; ++qi)
#pragma unroll
            for (int kt = 0; kt < 2; ++kt) {
                v8f c = zero;
#pragma unroll
                for (int hd = 0; hd < 2; ++hd)
                    c = __builtin_amdgcn_wmma_f32_16x16x32_bf16(
                        false, kf[kt][hd].v, false, qf[qi][hd].v, (short)0, c, false, false);
                st[qi][kt] = c;
            }

        // additive mask bias for this lane's 16 keys (vector loads)
        union { float4 f4[4]; float f[16]; } mbu;
#pragma unroll
        for (int kt = 0; kt < 2; ++kt) {
            const float* mp = mrow + key0 + kt * 16 + hi * 8;
            mbu.f4[kt * 2]     = *(const float4*)(mp);
            mbu.f4[kt * 2 + 1] = *(const float4*)(mp + 4);
        }

        // per-q-tile online softmax + P^T -> LDS
#pragma unroll
        for (int qi = 0; qi < 2; ++qi) {
            float sv[16];
            float cmax = -1e30f;
#pragma unroll
            for (int kt = 0; kt < 2; ++kt)
#pragma unroll
                for (int r = 0; r < 8; ++r) {
                    const float s = st[qi][kt][r] * 0.125f + mbu.f[kt * 8 + r];
                    sv[kt * 8 + r] = s;
                    cmax = fmaxf(cmax, s);
                }
            cmax = fmaxf(cmax, __shfl_xor(cmax, 16, 32));
            const float mnew  = fmaxf(mrun[qi], cmax);
            const float alpha = __expf(mrun[qi] - mnew);
            float psum = 0.f;
#pragma unroll
            for (int i = 0; i < 16; ++i) {
                sv[i] = __expf(sv[i] - mnew);
                psum += sv[i];
            }
            psum += __shfl_xor(psum, 16, 32);
            lsum[qi] = lsum[qi] * alpha + psum;
            mrun[qi] = mnew;

            // rescale O accumulators by per-query alpha (broadcast into C rows)
            float av[8];
#pragma unroll
            for (int r = 0; r < 8; ++r) av[r] = __shfl(alpha, hi * 8 + r, 32);
#pragma unroll
            for (int nt = 0; nt < 4; ++nt)
#pragma unroll
                for (int r = 0; r < 8; ++r) o[qi][nt][r] *= av[r];

            // P^T -> LDS as [q][kk] bf16 (pairs packed into dwords)
            unsigned int* pq = pw + qi * 256;
#pragma unroll
            for (int kt = 0; kt < 2; ++kt)
#pragma unroll
                for (int r = 0; r < 8; r += 2) {
                    const int kk = kt * 16 + hi * 8 + r;
                    const unsigned int pk =
                        (unsigned int)f32_to_bf16u(sv[kt * 8 + r]) |
                        ((unsigned int)f32_to_bf16u(sv[kt * 8 + r + 1]) << 16);
                    pq[l16 * 16 + (kk >> 1)] = pk;
                }
        }
        asm volatile("s_wait_dscnt 0" ::: "memory");

        // P as WMMA A-fragments (q x 32 keys), one per q-tile
        Frag pf[2];
#pragma unroll
        for (int qi = 0; qi < 2; ++qi) {
            const unsigned short* pp =
                (const unsigned short*)(pw + qi * 256) + l16 * 32 + hi * 8;
            pf[qi].u4[0] = *(const uint4*)(pp);
            pf[qi].u4[1] = *(const uint4*)(pp + 16);
        }
        // O += P x V: each V fragment feeds both q-tiles (8 WMMAs)
#pragma unroll
        for (int nt = 0; nt < 4; ++nt) {
            Frag vf;
            const unsigned short* vp = vptr + (size_t)(nt * 16 + l16) * SQ + key0 + hi * 8;
            vf.u4[0] = *(const uint4*)(vp);
            vf.u4[1] = *(const uint4*)(vp + 16);
#pragma unroll
            for (int qi = 0; qi < 2; ++qi)
                o[qi][nt] = __builtin_amdgcn_wmma_f32_16x16x32_bf16(
                    false, pf[qi].v, false, vf.v, (short)0, o[qi][nt], false, false);
        }
    }

    // normalize by l and store ctx [B,S,D] bf16
#pragma unroll
    for (int qi = 0; qi < 2; ++qi) {
        const float linv = 1.0f / lsum[qi];
        float lv[8];
#pragma unroll
        for (int r = 0; r < 8; ++r) lv[r] = __shfl(linv, hi * 8 + r, 32);
#pragma unroll
        for (int nt = 0; nt < 4; ++nt) {
            const int col = h * DK + nt * 16 + l16;
#pragma unroll
            for (int r = 0; r < 8; ++r) {
                const int q = qt * 32 + qi * 16 + hi * 8 + r;
                ctx[(size_t)(b * SQ + q) * DM + col] = f32_to_bf16u(o[qi][nt][r] * lv[r]);
            }
        }
    }
}

// ---------------------------------------------------------------------------
// launch
// ---------------------------------------------------------------------------
extern "C" void kernel_launch(void* const* d_in, const int* in_sizes, int n_in,
                              void* d_out, int out_size, void* d_ws, size_t ws_size,
                              hipStream_t stream) {
    (void)in_sizes; (void)n_in; (void)out_size; (void)ws_size;
    const float* x     = (const float*)d_in[0];
    const int*   mask  = (const int*)d_in[1];
    const float* Wq    = (const float*)d_in[2];
    const float* bq    = (const float*)d_in[3];
    const float* Wk    = (const float*)d_in[4];
    const float* bk    = (const float*)d_in[5];
    const float* Wv    = (const float*)d_in[6];
    const float* bv    = (const float*)d_in[7];
    const float* Wo    = (const float*)d_in[8];
    const float* bo    = (const float*)d_in[9];
    const float* W1    = (const float*)d_in[10];
    const float* b1    = (const float*)d_in[11];
    const float* W2    = (const float*)d_in[12];
    const float* b2    = (const float*)d_in[13];
    const float* al1   = (const float*)d_in[14];
    const float* bi1   = (const float*)d_in[15];
    const float* al2   = (const float*)d_in[16];
    const float* bi2   = (const float*)d_in[17];

    // workspace carve-up (256B aligned slabs)
    char* ws = (char*)d_ws;
    size_t off = 0;
    auto carve = [&](size_t bytes) {
        char* p = ws + off;
        off += (bytes + 255) & ~(size_t)255;
        return p;
    };
    unsigned short* hb   = (unsigned short*)carve((size_t)MT * DM * 2);
    unsigned short* WqT  = (unsigned short*)carve((size_t)DM * DM * 2);
    unsigned short* WkT  = (unsigned short*)carve((size_t)DM * DM * 2);
    unsigned short* WvT  = (unsigned short*)carve((size_t)DM * DM * 2);
    unsigned short* WoT  = (unsigned short*)carve((size_t)DM * DM * 2);
    unsigned short* W1T  = (unsigned short*)carve((size_t)DM * FF * 2);
    unsigned short* W2T  = (unsigned short*)carve((size_t)FF * DM * 2);
    unsigned short* qb   = (unsigned short*)carve((size_t)MT * DM * 2);
    unsigned short* kb   = (unsigned short*)carve((size_t)MT * DM * 2);
    unsigned short* vtb  = (unsigned short*)carve((size_t)MT * DM * 2);
    unsigned short* ctxb = (unsigned short*)carve((size_t)MT * DM * 2);
    float*          x1   = (float*)carve((size_t)MT * DM * 4);
    unsigned short* h2   = (unsigned short*)carve((size_t)MT * DM * 2);
    unsigned short* ffb  = (unsigned short*)carve((size_t)MT * FF * 2);
    float*          mb   = (float*)carve((size_t)BB * SQ * 4);

    // 1) weight convert+transpose fp32 -> bf16 [N,K]; mask -> additive bias
    wtrans_kernel<<<DM * DM / 256, 256, 0, stream>>>(Wq, WqT, DM, DM, 10);
    wtrans_kernel<<<DM * DM / 256, 256, 0, stream>>>(Wk, WkT, DM, DM, 10);
    wtrans_kernel<<<DM * DM / 256, 256, 0, stream>>>(Wv, WvT, DM, DM, 10);
    wtrans_kernel<<<DM * DM / 256, 256, 0, stream>>>(Wo, WoT, DM, DM, 10);
    wtrans_kernel<<<DM * FF / 256, 256, 0, stream>>>(W1, W1T, DM, FF, 10);  // [FF,DM]
    wtrans_kernel<<<FF * DM / 256, 256, 0, stream>>>(W2, W2T, FF, DM, 12);  // [DM,FF]
    maskbias_kernel<<<BB * SQ / 256, 256, 0, stream>>>(mask, mb);

    // 2) pre-norm 1
    ln_kernel<<<MT, 256, 0, stream>>>(x, al1, bi1, hb);

    // 3) QKV projections: block tile 128x128
    dim3 g1(DM / 128, MT / 128);
    gemm_kernel<EPI_BF16><<<g1, 256, 0, stream>>>(hb, WqT, bq, nullptr, qb, DM, DM);
    gemm_kernel<EPI_BF16><<<g1, 256, 0, stream>>>(hb, WkT, bk, nullptr, kb, DM, DM);
    gemm_kernel<EPI_VT><<<g1, 256, 0, stream>>>(hb, WvT, bv, nullptr, vtb, DM, DM);

    // 4) attention: BB*NH*(SQ/32) = 2048 waves = 512 blocks of 4 waves
    attn_kernel<<<BB * NH * (SQ / 32) / 4, 128, 0, stream>>>(qb, kb, vtb, mb, ctxb);

    // 5) output projection + residual -> x1 (fp32)
    gemm_kernel<EPI_F32_RES><<<g1, 256, 0, stream>>>(ctxb, WoT, bo, x, x1, DM, DM);

    // 6) pre-norm 2
    ln_kernel<<<MT, 256, 0, stream>>>(x1, al2, bi2, h2);

    // 7) FFN up + ReLU (bf16)
    dim3 g2(FF / 128, MT / 128);
    gemm_kernel<EPI_BF16_RELU><<<g2, 256, 0, stream>>>(h2, W1T, b1, nullptr, ffb, FF, DM);

    // 8) FFN down + residual -> d_out (fp32)
    gemm_kernel<EPI_F32_RES><<<g1, 256, 0, stream>>>(ffb, W2T, b2, x1, (float*)d_out, DM, FF);
}
